// LstmLayer_80925773791510
// MI455X (gfx1250) — compile-verified, matrix operand order checked by reading, not run on hardware
//
#include <hip/hip_runtime.h>

// LSTM layer for MI455X (gfx1250, wave32, WMMA).
//  - bf16 multiplies / fp32 accumulate via v_wmma_f32_16x16x32_bf16.
//  - Weights pre-swizzled ONCE into WMMA B-fragment layout (16 MB bf16, L2-resident).
//  - One fused kernel per timestep; each wave computes ALL FOUR gate tiles of a
//    16x16 (batch x hidden) block over a K-chunk (A-fragment reused across 4 WMMAs,
//    4 independent accumulator chains for XDL ILP). 8 waves/WG split K=2048
//    (2048 waves chip-wide so co-resident waves hide each other's load waits),
//    partials reduced through LDS, then fused sigmoid/tanh cell update.
//  - Fragment streams have constant 1 KB stride -> fully unrolled K-loop with
//    immediate-offset global_load_b128.
//  - h written straight into next step's pre-swizzled A-fragment buffer;
//    x[t+1] converted off the critical path by step t's kernel.

typedef __attribute__((ext_vector_type(16))) __bf16 v16bf;
typedef __attribute__((ext_vector_type(8)))  float  v8f;

static constexpr int SEQ   = 512;
static constexpr int BATCH = 64;
static constexpr int IDIM  = 1024;
static constexpr int HDIM  = 1024;
static constexpr int GDIM  = 4 * HDIM;   // 4096 gate columns

// ---- WMMA 16x32 bf16 fragment layout (CDNA5 ISA §7.12.2) ----
// Within a 16x32 tile, element (m, k) lives at:
//   lane = (m&15) + ((k%16)>=8 ? 16 : 0)
//   vreg v = (k>=16 ? 4 : 0) + ((k&7)>>1),  half = k&1
// Tiles stored fragment-major: tile*512 + lane*16 + v*2 + half  (bf16 elems)
__device__ __forceinline__ int frag_pos(int m, int k) {
  int lane = (m & 15) + (((k & 15) >= 8) ? 16 : 0);
  int v    = ((k & 16) ? 4 : 0) + ((k & 7) >> 1);
  return lane * 16 + v * 2 + (k & 1);
}

// A-fragment buffer for a [64 x 1024] activation matrix:
// tiles indexed (bt = row/16 in 0..3, kt = k/32 in 0..31), 512 bf16 per tile.
__device__ __forceinline__ int a_off(int row, int k) {
  return ((row >> 4) * 32 + (k >> 5)) * 512 + frag_pos(row & 15, k & 31);
}

// ---------------- weight swizzle: [W_i ; W_h] fp32 -> bf16 B-fragments ---------
// Combined K = 2048 (rows 0..1023 = W_i, 1024..2047 = W_h), N = 4096.
// Tiles indexed (nt = col/16 in 0..255, kt = row/32 in 0..63), 512 bf16 each.
__global__ __launch_bounds__(256) void swizzle_weights(
    const float* __restrict__ wi, const float* __restrict__ wh,
    __bf16* __restrict__ Wt)
{
  int idx = blockIdx.x * 256 + threadIdx.x;    // 0 .. 2048*4096-1
  int r = idx >> 12;                           // combined K row
  int c = idx & 4095;                          // gate column
  float v = (r < IDIM) ? wi[(size_t)r * GDIM + c]
                       : wh[(size_t)(r - IDIM) * GDIM + c];
  Wt[(size_t)(((c >> 4) * 64 + (r >> 5)) * 512 + frag_pos(c & 15, r & 31))] = (__bf16)v;
}

// ---------------- init: c state, h0 + x[0] into A-fragment buffers -------------
__global__ __launch_bounds__(256) void init_state(
    const float* __restrict__ x0, const float* __restrict__ h0,
    const float* __restrict__ c0, __bf16* __restrict__ xf,
    __bf16* __restrict__ hf, float* __restrict__ cws)
{
  int idx = blockIdx.x * 256 + threadIdx.x;    // 0 .. 65535  (B*H == B*I)
  int b = idx >> 10, j = idx & 1023;
  cws[idx] = c0[idx];
  hf[a_off(b, j)] = (__bf16)h0[idx];
  xf[a_off(b, j)] = (__bf16)x0[idx];
}

// ---------------- fused LSTM step ---------------------------------------------
// grid = 256 workgroups (bt in 0..3, jt in 0..63), block = 256 (8 waves).
// Wave w handles combined-K chunk [w*256, (w+1)*256): waves 0-3 cover x@W_i,
// waves 4-7 cover h@W_h. Each wave computes all 4 gate tiles (A reused 4x).
__global__ __launch_bounds__(256) void lstm_step(
    const __bf16* __restrict__ xr,  // A-fragments of x[t]
    const __bf16* __restrict__ hr,  // A-fragments of h[t-1]
    __bf16*       __restrict__ hw,  // A-fragments of h[t]   (written here)
    const __bf16* __restrict__ Wt,  // B-fragments of [W_i;W_h]
    const float*  __restrict__ bias,
    float*        __restrict__ cws, // cell state [B,H] fp32
    float*        __restrict__ out, // outputs [S,B,H]
    const float*  __restrict__ xin, // raw inputs [S,B,I] fp32
    __bf16*       __restrict__ xw,  // A-fragments of x[t+1] (written here)
    int t)
{
  // K-partial gate sums: [k-chunk wave][gate][lane*8 + r]  (32 KB)
  __shared__ float part[8][4][256];

  const int tid  = threadIdx.x;
  const int w    = tid >> 5;          // K-chunk index 0..7
  const int lane = tid & 31;
  const int bt   = blockIdx.x >> 6;   // batch tile 0..3
  const int jt   = blockIdx.x & 63;   // hidden tile 0..63

  v8f acc[4] = {v8f{}, v8f{}, v8f{}, v8f{}};
  if (w == 0) {
    // bias folded once, into K-chunk 0's accumulators
    int col = jt * 16 + (lane & 15);
#pragma unroll
    for (int g = 0; g < 4; ++g) {
      float bv = bias[g * HDIM + col];
      acc[g] = v8f{bv, bv, bv, bv, bv, bv, bv, bv};
    }
  }

  const int ktl_base = (w & 3) * 8;   // tile index within x- or h-fragment buffer
  const int ktg_base = w * 8;         // tile index within combined-K weight tiles

  // One base pointer per stream; k-step stride is 32 v16bf = 1024 bytes, so the
  // fully unrolled loop uses constant immediate offsets (kk*1024) on each load.
  const v16bf* pa  = (const v16bf*)((w < 4) ? xr : hr)
                     + (size_t)(bt * 32 + ktl_base) * 32 + lane;
  const v16bf* pb0 = (const v16bf*)Wt + (size_t)((0 * 64 + jt) * 64 + ktg_base) * 32 + lane;
  const v16bf* pb1 = (const v16bf*)Wt + (size_t)((1 * 64 + jt) * 64 + ktg_base) * 32 + lane;
  const v16bf* pb2 = (const v16bf*)Wt + (size_t)((2 * 64 + jt) * 64 + ktg_base) * 32 + lane;
  const v16bf* pb3 = (const v16bf*)Wt + (size_t)((3 * 64 + jt) * 64 + ktg_base) * 32 + lane;

#pragma unroll
  for (int kk = 0; kk < 8; ++kk) {
    v16bf a  = pa [kk * 32];
    v16bf b0 = pb0[kk * 32];
    v16bf b1 = pb1[kk * 32];
    v16bf b2 = pb2[kk * 32];
    v16bf b3 = pb3[kk * 32];
    // warm near caches for the B streams 4 k-steps ahead (compile-time guard)
    if (kk < 4) {
      __builtin_prefetch((const void*)(pb0 + (kk + 4) * 32), 0, 3);
      __builtin_prefetch((const void*)(pb2 + (kk + 4) * 32), 0, 3);
    }
    acc[0] = __builtin_amdgcn_wmma_f32_16x16x32_bf16(false, a, false, b0,
                                                     (short)0, acc[0], false, false);
    acc[1] = __builtin_amdgcn_wmma_f32_16x16x32_bf16(false, a, false, b1,
                                                     (short)0, acc[1], false, false);
    acc[2] = __builtin_amdgcn_wmma_f32_16x16x32_bf16(false, a, false, b2,
                                                     (short)0, acc[2], false, false);
    acc[3] = __builtin_amdgcn_wmma_f32_16x16x32_bf16(false, a, false, b3,
                                                     (short)0, acc[3], false, false);
  }

  // lane-contiguous partial store (8 consecutive floats per lane -> b128 merges)
#pragma unroll
  for (int g = 0; g < 4; ++g)
#pragma unroll
    for (int r = 0; r < 8; ++r)
      part[w][g][lane * 8 + r] = acc[g][r];
  __syncthreads();

  // fused pointwise: reduce 8 K-partials, sigmoid/tanh, cell update, outputs +
  // next-step h fragments. Element e = m*16+n of the 16x16 (batch x hidden) tile.
  {
    int e = tid;                                       // 256 threads, 256 elems
    int m = e >> 4, n = e & 15;
    int li = (n + ((m & 8) ? 16 : 0)) * 8 + (m & 7);   // C/D layout inverse
    float gi = 0.f, gf = 0.f, gg = 0.f, go = 0.f;
#pragma unroll
    for (int p = 0; p < 8; ++p) {
      gi += part[p][0][li];
      gf += part[p][1][li];
      gg += part[p][2][li];
      go += part[p][3][li];
    }
    float ig = 1.f / (1.f + __expf(-gi));
    float fg = 1.f / (1.f + __expf(-gf));
    float cg = tanhf(gg);
    float og = 1.f / (1.f + __expf(-go));
    int b = bt * 16 + m;
    int j = jt * 16 + n;
    int idx = b * HDIM + j;
    float cn = fg * cws[idx] + ig * cg;
    cws[idx] = cn;
    float hn = og * tanhf(cn);
    out[(size_t)t * BATCH * HDIM + idx] = hn;
    hw[a_off(b, j)] = (__bf16)hn;
  }

  // convert x[t+1] into next step's A-fragment buffer (off the critical path):
  // 256 wgs * 256 threads = 65536 threads, exactly one element each.
  if (t + 1 < SEQ) {
    int e = blockIdx.x * 256 + tid;
    int b = e >> 10, i = e & 1023;
    xw[a_off(b, i)] = (__bf16)xin[(size_t)(t + 1) * BATCH * IDIM + e];
  }
}

// ---------------- finalize: h_T and c_T appended to outputs --------------------
__global__ __launch_bounds__(256) void finalize(float* __restrict__ out,
                                                const float* __restrict__ cws)
{
  int idx = blockIdx.x * 256 + threadIdx.x;     // 0 .. 65535
  const size_t base = (size_t)SEQ * BATCH * HDIM;
  out[base + idx]                = out[(size_t)(SEQ - 1) * BATCH * HDIM + idx];
  out[base + BATCH * HDIM + idx] = cws[idx];
}

extern "C" void kernel_launch(void* const* d_in, const int* in_sizes, int n_in,
                              void* d_out, int out_size, void* d_ws, size_t ws_size,
                              hipStream_t stream) {
  const float* x    = (const float*)d_in[0];
  const float* h0   = (const float*)d_in[1];
  const float* c0   = (const float*)d_in[2];
  const float* wi   = (const float*)d_in[3];
  const float* wh   = (const float*)d_in[4];
  const float* bias = (const float*)d_in[5];
  float* out = (float*)d_out;

  // workspace layout:
  //   Wt     : 256*64*512 bf16 = 8,388,608  (16 MB)
  //   xf[2]  : 2 * 65536 bf16   (x A-fragments, ping-pong)
  //   hf[2]  : 2 * 65536 bf16   (h A-fragments, ping-pong)
  //   cws    : 65536 fp32       (cell state)
  __bf16* Wt  = (__bf16*)d_ws;
  __bf16* xf0 = Wt + (size_t)8388608;
  __bf16* xf1 = xf0 + 65536;
  __bf16* hf0 = xf1 + 65536;
  __bf16* hf1 = hf0 + 65536;
  float*  cws = (float*)(hf1 + 65536);

  swizzle_weights<<<dim3(32768), dim3(256), 0, stream>>>(wi, wh, Wt);
  init_state<<<dim3(256), dim3(256), 0, stream>>>(x, h0, c0, xf0, hf0, cws);

  for (int t = 0; t < SEQ; ++t) {
    const bool p = (t & 1) != 0;
    const __bf16* xr = p ? xf1 : xf0;
    __bf16*       xw = p ? xf0 : xf1;
    const __bf16* hr = p ? hf1 : hf0;
    __bf16*       hw = p ? hf0 : hf1;
    lstm_step<<<dim3(256), dim3(256), 0, stream>>>(xr, hr, hw, Wt, bias, cws,
                                                   out, x, xw, t);
  }

  finalize<<<dim3(256), dim3(256), 0, stream>>>(out, cws);
}
